// RLAAttention_56710748176767
// MI455X (gfx1250) — matrix operand order repaired and checked by reading, not compiled, not run on hardware
//
#include <hip/hip_runtime.h>
#include <hip/hip_bf16.h>

// ---------------------------------------------------------------------------
// Types
// ---------------------------------------------------------------------------
typedef __bf16 bf16;
typedef __bf16 v16bf __attribute__((ext_vector_type(16)));
typedef __bf16 v8bf  __attribute__((ext_vector_type(8)));
typedef float  v8f   __attribute__((ext_vector_type(8)));
typedef float  v4f   __attribute__((ext_vector_type(4)));
typedef int    v4i   __attribute__((ext_vector_type(4)));

#define B_     2
#define S_     2048
#define D_     2048
#define NH_    16
#define NKV_   4
#define HD_    128
#define QDIM_  2048
#define KVDIM_ 512
#define QKVN_  3072
#define MTOT_  (B_ * S_)   // 4096

// gfx1250 async global->LDS path (ASYNCcnt), guarded so the file still
// compiles on toolchains (and the host pass) that lack the builtin.
#if defined(__has_builtin)
#if __has_builtin(__builtin_amdgcn_global_load_async_to_lds_b128)
#define HAVE_ASYNC_LDS 1
#endif
#endif

#if defined(HAVE_ASYNC_LDS) && defined(__AMDGCN__)
// Builtin prototype (from clang diagnostic): (int4 AS1*, int4 AS3*, Ii, Ii)
typedef __attribute__((address_space(1))) v4i* as1_v4i;
typedef __attribute__((address_space(3))) v4i* as3_v4i;
#define ASYNC_CP16(gsrc, ldst)                                        \
  __builtin_amdgcn_global_load_async_to_lds_b128((as1_v4i)(gsrc),     \
                                                 (as3_v4i)(ldst), 0, 0)
#endif

// ---------------------------------------------------------------------------
// fp32 -> bf16 conversion (vectorized: 2x b128 load, 1x b128 store)
// ---------------------------------------------------------------------------
__global__ __launch_bounds__(256) void cvt_bf16_kernel(
    const float* __restrict__ src, bf16* __restrict__ dst, int n) {
  int i = (blockIdx.x * 256 + threadIdx.x) * 8;
  if (i < n) {
    v4f f0 = *(const v4f*)(src + i);
    v4f f1 = *(const v4f*)(src + i + 4);
    v8bf o;
    o[0] = (bf16)f0[0]; o[1] = (bf16)f0[1]; o[2] = (bf16)f0[2];
    o[3] = (bf16)f0[3]; o[4] = (bf16)f1[0]; o[5] = (bf16)f1[1];
    o[6] = (bf16)f1[2]; o[7] = (bf16)f1[3];
    *(v8bf*)(dst + i) = o;
  }
}

// ---------------------------------------------------------------------------
// bf16 WMMA GEMM:  C[M,N] (f32) = A[M,K] (bf16, row-major) * Bw[N,K]^T (bf16)
// Block: 256 threads = 8 waves. Wave tile: 32 rows x 64 cols
// (2 A-frags x 4 B-frags -> 8 v_wmma per 32-deep K step; each B fragment
// feeds two WMMAs). Block tile: 256 rows x 64 cols.
// ---------------------------------------------------------------------------
__global__ __launch_bounds__(256) void gemm_bf16_wmma_kernel(
    const bf16* __restrict__ A, const bf16* __restrict__ Bw,
    float* __restrict__ Cf, int M, int N, int K) {
  const int lane = threadIdx.x & 31;
  const int wave = threadIdx.x >> 5;
  const int hh = lane >> 4;    // K-half selector (wave32 A/B layout)
  const int ll = lane & 15;
  const int m0 = blockIdx.y * 256 + wave * 32;
  const int n0 = blockIdx.x * 64;

  const bf16* __restrict__ arow0 = A + (size_t)(m0 + ll) * K;
  const bf16* __restrict__ arow1 = A + (size_t)(m0 + 16 + ll) * K;

  v8f acc[2][4] = {};
  for (int k = 0; k < K; k += 32) {
    __builtin_prefetch(arow0 + k + 256, 0, 0);  // global_prefetch ahead in K
    __builtin_prefetch(arow1 + k + 256, 0, 0);
    // A fragments 16x32: lane row = ll; elems i -> K = k+16*hh+i
    v8bf a0lo = *(const v8bf*)(arow0 + k + 16 * hh);
    v8bf a0hi = *(const v8bf*)(arow0 + k + 16 * hh + 8);
    v8bf a1lo = *(const v8bf*)(arow1 + k + 16 * hh);
    v8bf a1hi = *(const v8bf*)(arow1 + k + 16 * hh + 8);
    v16bf a0 = __builtin_shufflevector(a0lo, a0hi, 0, 1, 2, 3, 4, 5, 6, 7, 8,
                                       9, 10, 11, 12, 13, 14, 15);
    v16bf a1 = __builtin_shufflevector(a1lo, a1hi, 0, 1, 2, 3, 4, 5, 6, 7, 8,
                                       9, 10, 11, 12, 13, 14, 15);
#pragma unroll
    for (int j = 0; j < 4; ++j) {
      // B fragment 32x16: lane col = ll; elems i -> K = k + 16*hh + i
      const bf16* bp = Bw + (size_t)(n0 + j * 16 + ll) * K + k + 16 * hh;
      v16bf bfv = *(const v16bf*)bp;
      acc[0][j] = __builtin_amdgcn_wmma_f32_16x16x32_bf16(
          false, a0, false, bfv, (short)0, acc[0][j], false, false);
      acc[1][j] = __builtin_amdgcn_wmma_f32_16x16x32_bf16(
          false, a1, false, bfv, (short)0, acc[1][j], false, false);
    }
  }
  // C layout: lane col = ll, row = v + 8*hh
#pragma unroll
  for (int mi = 0; mi < 2; ++mi) {
#pragma unroll
    for (int j = 0; j < 4; ++j) {
#pragma unroll
      for (int v = 0; v < 8; ++v) {
        Cf[(size_t)(m0 + mi * 16 + v + 8 * hh) * N + (n0 + j * 16 + ll)] =
            acc[mi][j][v];
      }
    }
  }
}

// ---------------------------------------------------------------------------
// RoPE + pack: reads xqkv fp32 [B*S, 3072], writes
//   qw  [B,NH ,S,HD] bf16   (rotated, pre-scaled by 1/sqrt(HD)*log2(e))
//   kw  [B,NKV,S,HD] bf16   (rotated)
//   vtw [B,NKV,HD,S] bf16   (transposed so V is a direct WMMA B-operand)
// ---------------------------------------------------------------------------
__global__ __launch_bounds__(256) void rope_pack_kernel(
    const float* __restrict__ xqkv, const float* __restrict__ freqs,
    bf16* __restrict__ qw, bf16* __restrict__ kw, bf16* __restrict__ vtw) {
  const int bs = blockIdx.x;
  const int b = bs / S_, s = bs - b * S_;
  const int t = threadIdx.x;
  const float* __restrict__ row = xqkv + (size_t)bs * QKVN_;
  const float* __restrict__ fr = freqs + (size_t)s * HD_;
  // 1/sqrt(128) * log2(e): softmax scale folded into Q, base-2 exponentials.
  const float qscale = 0.08838834764831845f * 1.4426950408889634f;

  for (int p = t; p < QDIM_ / 2; p += 256) {
    int h = p >> 6, i = p & 63;
    float t0 = row[h * HD_ + 2 * i], t1 = row[h * HD_ + 2 * i + 1];
    float f0 = fr[2 * i], f1 = fr[2 * i + 1];
    size_t base = (((size_t)(b * NH_ + h)) * S_ + s) * HD_;
    qw[base + 2 * i]     = (bf16)((t0 * f1 - t1 * f0) * qscale);
    qw[base + 2 * i + 1] = (bf16)((t0 * f0 + t1 * f1) * qscale);
  }
  for (int p = t; p < KVDIM_ / 2; p += 256) {
    int kv = p >> 6, i = p & 63;
    float t0 = row[QDIM_ + kv * HD_ + 2 * i];
    float t1 = row[QDIM_ + kv * HD_ + 2 * i + 1];
    float f0 = fr[2 * i], f1 = fr[2 * i + 1];
    size_t base = (((size_t)(b * NKV_ + kv)) * S_ + s) * HD_;
    kw[base + 2 * i]     = (bf16)(t0 * f1 - t1 * f0);
    kw[base + 2 * i + 1] = (bf16)(t0 * f0 + t1 * f1);
  }
  for (int e = t; e < KVDIM_; e += 256) {
    int kv = e >> 7, d = e & 127;
    vtw[(((size_t)(b * NKV_ + kv)) * HD_ + d) * S_ + s] =
        (bf16)row[QDIM_ + KVDIM_ + e];
  }
}

// ---------------------------------------------------------------------------
// Causal flash attention, GQA (4 Q heads per KV head).
// Grid: (S/64, NH, B). Block: 128 threads = 4 waves; each wave owns 16 Q rows.
// K/V^T tiles (32 keys) staged in LDS via async global->LDS DMA when
// available; online softmax with 16-lane butterfly reductions; P transposed
// C->A layout via per-wave LDS bounce fenced with s_wait_dscnt.
// ---------------------------------------------------------------------------
__global__ __launch_bounds__(128) void flash_attn_kernel(
    const bf16* __restrict__ qw, const bf16* __restrict__ kw,
    const bf16* __restrict__ vtw, bf16* __restrict__ attn) {
  __shared__ __align__(128) bf16 lds_k[32][128];   // [key][d]
  __shared__ __align__(128) bf16 lds_v[128][32];   // [d][key]  (V^T)
  __shared__ __align__(128) bf16 lds_p[4][16][32]; // per-wave P bounce

  const int q0 = blockIdx.x * 64;
  const int h = blockIdx.y;
  const int b = blockIdx.z;
  const int kvh = h >> 2;  // GQA: 4 Q heads share one KV head
  const int tid = threadIdx.x;
  const int wave = tid >> 5, lane = tid & 31;
  const int hh = lane >> 4, ll = lane & 15;
  const int qrow0 = q0 + wave * 16;

  // Q fragments for this wave's 16 rows, all of HD=128 (4 x 16x32 A-frags)
  const bf16* __restrict__ qbase =
      qw + (((size_t)(b * NH_ + h)) * S_ + (qrow0 + ll)) * HD_;
  v16bf qf[4];
#pragma unroll
  for (int c = 0; c < 4; ++c) {
    v8bf lo = *(const v8bf*)(qbase + c * 32 + 16 * hh);
    v8bf hi = *(const v8bf*)(qbase + c * 32 + 16 * hh + 8);
    qf[c] = __builtin_shufflevector(lo, hi, 0, 1, 2, 3, 4, 5, 6, 7, 8, 9, 10,
                                    11, 12, 13, 14, 15);
  }

  v8f acc[8] = {};          // output [16 rows x 128 d] as 8 C-tiles
  float mrow[8], lrow[8];   // per-lane row stats (row = v + 8*hh)
#pragma unroll
  for (int v = 0; v < 8; ++v) { mrow[v] = -1e30f; lrow[v] = 0.0f; }

  const bf16* __restrict__ kbase = kw + ((size_t)(b * NKV_ + kvh)) * S_ * HD_;
  const bf16* __restrict__ vbase = vtw + ((size_t)(b * NKV_ + kvh)) * HD_ * S_;

  for (int kb = 0; kb < q0 + 64; kb += 32) {
    // Cooperative tile fill: K tile 32x128 (8 KB) + V^T tile 128x32 (8 KB).
    {
      int r = tid >> 2, doff = (tid & 3) * 32;  // K tile row/col chunk
      const bf16* gk = kbase + (size_t)(kb + r) * HD_ + doff;
      const bf16* gv = vbase + (size_t)tid * S_ + kb;
#if defined(HAVE_ASYNC_LDS) && defined(__AMDGCN__)
      ASYNC_CP16(gk,      &lds_k[r][doff]);
      ASYNC_CP16(gk + 8,  &lds_k[r][doff + 8]);
      ASYNC_CP16(gk + 16, &lds_k[r][doff + 16]);
      ASYNC_CP16(gk + 24, &lds_k[r][doff + 24]);
      ASYNC_CP16(gv,      &lds_v[tid][0]);
      ASYNC_CP16(gv + 8,  &lds_v[tid][8]);
      ASYNC_CP16(gv + 16, &lds_v[tid][16]);
      ASYNC_CP16(gv + 24, &lds_v[tid][24]);
#if __has_builtin(__builtin_amdgcn_s_wait_asynccnt)
      __builtin_amdgcn_s_wait_asynccnt(0);
#else
      asm volatile("s_wait_asynccnt 0" ::: "memory");
#endif
#else
      *(v16bf*)&lds_k[r][doff]      = *(const v16bf*)gk;
      *(v16bf*)&lds_k[r][doff + 16] = *(const v16bf*)(gk + 16);
      *(v16bf*)&lds_v[tid][0]       = *(const v16bf*)gv;
      *(v16bf*)&lds_v[tid][16]      = *(const v16bf*)(gv + 16);
#endif
    }
    __syncthreads();

    // Scores: two 16-key tiles, contract over HD=128 (4 wmma each)
    v8f sc[2] = {};
#pragma unroll
    for (int jt = 0; jt < 2; ++jt) {
#pragma unroll
      for (int c = 0; c < 4; ++c) {
        v16bf kf = *(const v16bf*)&lds_k[jt * 16 + ll][c * 32 + 16 * hh];
        sc[jt] = __builtin_amdgcn_wmma_f32_16x16x32_bf16(
            false, qf[c], false, kf, (short)0, sc[jt], false, false);
      }
    }

    // Causal mask + online softmax update (base-2; scale folded into Q)
#pragma unroll
    for (int v = 0; v < 8; ++v) {
      int row = qrow0 + v + 8 * hh;
      float s0 = (kb + ll      > row) ? -1e30f : sc[0][v];
      float s1 = (kb + 16 + ll > row) ? -1e30f : sc[1][v];
      float mx = fmaxf(s0, s1);
#pragma unroll
      for (int off = 8; off > 0; off >>= 1)
        mx = fmaxf(mx, __shfl_xor(mx, off, 16));
      float mnew = fmaxf(mrow[v], mx);
      float p0 = exp2f(s0 - mnew);
      float p1 = exp2f(s1 - mnew);
      float corr = exp2f(mrow[v] - mnew);
      float ps = p0 + p1;
#pragma unroll
      for (int off = 8; off > 0; off >>= 1) ps += __shfl_xor(ps, off, 16);
      lrow[v] = lrow[v] * corr + ps;
      mrow[v] = mnew;
#pragma unroll
      for (int dt = 0; dt < 8; ++dt) acc[dt][v] *= corr;
      lds_p[wave][v + 8 * hh][ll]      = (bf16)p0;  // C-layout -> LDS
      lds_p[wave][v + 8 * hh][16 + ll] = (bf16)p1;
    }
    // Wave-local LDS RAW fence before re-reading P in A-layout
    asm volatile("s_wait_dscnt 0" ::: "memory");

    // Reload P as a 16x32 A-fragment
    v8bf plo = *(const v8bf*)&lds_p[wave][ll][16 * hh];
    v8bf phi = *(const v8bf*)&lds_p[wave][ll][16 * hh + 8];
    v16bf pf = __builtin_shufflevector(plo, phi, 0, 1, 2, 3, 4, 5, 6, 7, 8, 9,
                                       10, 11, 12, 13, 14, 15);
    // PV: contract over 32 keys, 8 d-tiles of 16
#pragma unroll
    for (int dt = 0; dt < 8; ++dt) {
      v16bf vf = *(const v16bf*)&lds_v[dt * 16 + ll][16 * hh];
      acc[dt] = __builtin_amdgcn_wmma_f32_16x16x32_bf16(
          false, pf, false, vf, (short)0, acc[dt], false, false);
    }
    __syncthreads();  // tiles consumed before next iteration overwrites
  }

  // Normalize and emit bf16 [B*S, 2048] (col = h*128 + d) for the Wo GEMM
#pragma unroll
  for (int v = 0; v < 8; ++v) {
    float inv = 1.0f / lrow[v];
    int srow = qrow0 + v + 8 * hh;
    size_t rbase = ((size_t)(b * S_ + srow)) * D_ + h * HD_;
#pragma unroll
    for (int dt = 0; dt < 8; ++dt)
      attn[rbase + dt * 16 + ll] = (bf16)(acc[dt][v] * inv);
  }
}

// ---------------------------------------------------------------------------
// Host-side launcher
// ---------------------------------------------------------------------------
extern "C" void kernel_launch(void* const* d_in, const int* in_sizes, int n_in,
                              void* d_out, int out_size, void* d_ws,
                              size_t ws_size, hipStream_t stream) {
  const float* x     = (const float*)d_in[0];  // [B,S,D]
  const float* freqs = (const float*)d_in[1];  // [S,HD]
  const float* Wqkv  = (const float*)d_in[2];  // [3072, 2048]
  const float* Wo    = (const float*)d_in[3];  // [2048, 2048]
  float* out = (float*)d_out;                  // [B,S,D]

  char* ws = (char*)d_ws;
  bf16* xbf    = (bf16*)(ws);              // 16,777,216 B (reused as attn_bf)
  bf16* wqkvbf = (bf16*)(ws + 16777216);   // 12,582,912 B
  bf16* wobf   = (bf16*)(ws + 29360128);   //  8,388,608 B
  float* xqkv  = (float*)(ws + 37748736);  // 50,331,648 B
  bf16* qw     = (bf16*)(ws + 88080384);   // 16,777,216 B
  bf16* kw     = (bf16*)(ws + 104857600);  //  4,194,304 B
  bf16* vtw    = (bf16*)(ws + 109051904);  //  4,194,304 B -> 113,246,208 total
  bf16* attnbf = xbf;  // x no longer needed after the QKV GEMM

  // 1) bf16 conversions
  cvt_bf16_kernel<<<(MTOT_ * D_) / 2048, 256, 0, stream>>>(x, xbf,
                                                           MTOT_ * D_);
  cvt_bf16_kernel<<<(QKVN_ * D_) / 2048, 256, 0, stream>>>(Wqkv, wqkvbf,
                                                           QKVN_ * D_);
  cvt_bf16_kernel<<<(D_ * QDIM_) / 2048, 256, 0, stream>>>(Wo, wobf,
                                                           D_ * QDIM_);

  // 2) QKV projection: xqkv[4096,3072] = xbf * Wqkv^T (fp32 out for RoPE)
  gemm_bf16_wmma_kernel<<<dim3(QKVN_ / 64, MTOT_ / 256), 256, 0, stream>>>(
      xbf, wqkvbf, xqkv, MTOT_, QKVN_, D_);

  // 3) RoPE + pack into attention-friendly layouts
  rope_pack_kernel<<<MTOT_, 256, 0, stream>>>(xqkv, freqs, qw, kw, vtw);

  // 4) Causal GQA flash attention -> attnbf[4096, 2048]
  flash_attn_kernel<<<dim3(S_ / 64, NH_, B_), 128, 0, stream>>>(qw, kw, vtw,
                                                                attnbf);

  // 5) Output projection: out[4096,2048] = attnbf * Wo^T (fp32 out)
  gemm_bf16_wmma_kernel<<<dim3(D_ / 64, MTOT_ / 256), 256, 0, stream>>>(
      attnbf, wobf, out, MTOT_, D_, QDIM_);
}
